// get_model_85572928405978
// MI455X (gfx1250) — matrix operand order repaired and checked by reading, not compiled, not run on hardware
//
#include <hip/hip_runtime.h>
#include <hip/hip_bf16.h>

typedef __attribute__((ext_vector_type(16))) _Float16 v16h;
typedef __attribute__((ext_vector_type(8)))  _Float16 h8;
typedef __attribute__((ext_vector_type(8)))  float    v8f;

#define NB 2          // batch
#define N0 16000      // input points

// ---------------------------------------------------------------------------
// WMMA GEMM:  Y[M,N] = A[M,K] * X[K,N]  (+bias), f32 storage, f16 math,
// f32 accumulation via v_wmma_f32_16x16x32_f16.
// Block = 256 threads = 8 waves; each wave computes one 16x16 tile,
// block covers 128 rows x 16 cols. K staged in 64-wide chunks, 2 WMMA/chunk.
// LDS layouts are arranged so each lane's fragment = aligned 16B segments,
// giving ds_load_b128 fragment builds instead of scalar ds_load_u16.
// ---------------------------------------------------------------------------
__global__ __launch_bounds__(256)
void k_gemm(const float* __restrict__ A, const float* __restrict__ X,
            float* __restrict__ Y, int M, int K, int N,
            const float* __restrict__ bias) {
  // 72-half row stride = 144B: every 8-half (16B) segment stays 16B-aligned.
  __shared__ __align__(16) _Float16 sA[128][72];   // [row][k]  (k-chunk of 64)
  __shared__ __align__(16) _Float16 sBt[16][72];   // [n][k]    transposed B tile
  const int tid  = threadIdx.x;
  const int lane = tid & 31;
  const int wave = tid >> 5;
  const int laneHalf = lane >> 4;   // 0: lanes 0-15, 1: lanes 16-31
  const int lr = lane & 15;
  const int rowBase = blockIdx.y * 128;
  const int colBase = blockIdx.x * 16;
  const bool fullM = (rowBase + 128 <= M);

  v8f acc = {};

  for (int kk = 0; kk < K; kk += 64) {
    const bool fullK = (kk + 64 <= K);
    if (fullM && fullK) {
      // unguarded, coalesced staging (block-uniform fast path)
      for (int i = tid; i < 128 * 64; i += 256) {
        int r = i >> 6, c = i & 63;
        sA[r][c] = (_Float16)A[(size_t)(rowBase + r) * K + (kk + c)];
      }
      for (int i = tid; i < 64 * 16; i += 256) {
        int r = i >> 4, c = i & 15;
        sBt[c][r] = (_Float16)X[(size_t)(kk + r) * N + (colBase + c)];
      }
    } else {
      for (int i = tid; i < 128 * 64; i += 256) {
        int r = i >> 6, c = i & 63;
        int gr = rowBase + r, gc = kk + c;
        float v = (gr < M && gc < K) ? A[(size_t)gr * K + gc] : 0.0f;
        sA[r][c] = (_Float16)v;
      }
      for (int i = tid; i < 64 * 16; i += 256) {
        int r = i >> 4, c = i & 15;
        int gr = kk + r;   // columns always in range (N is a multiple of 16)
        float v = (gr < K) ? X[(size_t)gr * N + (colBase + c)] : 0.0f;
        sBt[c][r] = (_Float16)v;
      }
    }
    // prefetch next K-chunk while this one is consumed
    if (kk + 64 < K) {
      const float* pa = A + (size_t)(rowBase + (tid >> 1)) * K + (kk + 64) + (tid & 1) * 32;
      __builtin_prefetch(pa, 0, 1);
      const float* px = X + (size_t)(kk + 64 + (tid >> 4)) * N + colBase + (lr);
      __builtin_prefetch(px, 0, 1);
    }
    __syncthreads();

    const int arow = (wave << 4) + lr;
#pragma unroll
    for (int s = 0; s < 2; ++s) {
      // A 16x32 f16 layout: lanes<16 hold K {0..7,16..23}, lanes>=16 {8..15,24..31}
      const h8 a0 = *(const h8*)&sA[arow][s * 32 + 8 * laneHalf];
      const h8 a1 = *(const h8*)&sA[arow][s * 32 + 16 + 8 * laneHalf];
      // B 32x16 f16 layout: lanes<16 hold K {0..7,16..23}, lanes>=16 {8..15,24..31}, N=lane%16
      const h8 b0 = *(const h8*)&sBt[lr][s * 32 + 8 * laneHalf];
      const h8 b1 = *(const h8*)&sBt[lr][s * 32 + 16 + 8 * laneHalf];
      v16h af, bf;
#pragma unroll
      for (int h = 0; h < 8; ++h) {
        af[h] = a0[h]; af[h + 8] = a1[h];
        bf[h] = b0[h]; bf[h + 8] = b1[h];
      }
      acc = __builtin_amdgcn_wmma_f32_16x16x32_f16(false, af, false, bf,
                                                   (short)0, acc, false, false);
    }
    __syncthreads();
  }

  // D layout: VGPR r -> M = r + 8*laneHalf, N = lane%16
#pragma unroll
  for (int r = 0; r < 8; ++r) {
    int row = rowBase + (wave << 4) + r + (laneHalf ? 8 : 0);
    int col = colBase + lr;
    if (row < M && col < N) {
      float v = acc[r];
      if (bias) v += bias[row];
      Y[(size_t)row * N + col] = v;
    }
  }
}

// ---------------------------------------------------------------------------
// BatchNorm helpers (training-mode stats over all columns of each row)
// ---------------------------------------------------------------------------
__global__ __launch_bounds__(256)
void k_rowstats(const float* __restrict__ Y, int Ncols,
                float* __restrict__ mean, float* __restrict__ rstd) {
  __shared__ float ss[256], sq[256];
  int r = blockIdx.x;
  const float* row = Y + (size_t)r * Ncols;
  float s = 0.f, q = 0.f;
  for (int c = threadIdx.x; c < Ncols; c += 256) { float v = row[c]; s += v; q += v * v; }
  ss[threadIdx.x] = s; sq[threadIdx.x] = q;
  __syncthreads();
  for (int st = 128; st > 0; st >>= 1) {
    if (threadIdx.x < st) { ss[threadIdx.x] += ss[threadIdx.x + st]; sq[threadIdx.x] += sq[threadIdx.x + st]; }
    __syncthreads();
  }
  if (threadIdx.x == 0) {
    float mu = ss[0] / (float)Ncols;
    float var = sq[0] / (float)Ncols - mu * mu;
    mean[r] = mu;
    rstd[r] = rsqrtf(var + 1e-5f);
  }
}

__global__ __launch_bounds__(256)
void k_bn_relu(float* __restrict__ Y, long total, int Ncols,
               const float* __restrict__ mean, const float* __restrict__ rstd,
               const float* __restrict__ gamma, const float* __restrict__ beta) {
  long i = (long)blockIdx.x * 256 + threadIdx.x;
  if (i >= total) return;
  int r = (int)(i / Ncols);
  float v = (Y[i] - mean[r]) * rstd[r] * gamma[r] + beta[r];
  Y[i] = v > 0.f ? v : 0.f;
}

// ---------------------------------------------------------------------------
// Point-cloud kernels
// ---------------------------------------------------------------------------
__global__ __launch_bounds__(256)
void k_init_l0(const float* __restrict__ xyz, float* __restrict__ P, float* __restrict__ F) {
  int i = blockIdx.x * 256 + threadIdx.x;
  if (i >= NB * 6 * N0) return;
  int b = i / (6 * N0); int rem = i % (6 * N0); int c = rem / N0; int n = rem % N0;
  float v = xyz[i];
  F[(size_t)c * NB * N0 + (size_t)b * N0 + n] = v;
  if (c < 3) P[((size_t)b * N0 + n) * 3 + c] = v;
}

// Farthest point sampling: one block per batch, matches jax scan semantics
// (out[0]=0, then argmax after each min-update; first-index tie break).
__global__ __launch_bounds__(256)
void k_fps(const float* __restrict__ P, int Nin, int NPt,
           float* __restrict__ dist, int* __restrict__ outIdx) {
  int b = blockIdx.x;
  const float* p = P + (size_t)b * Nin * 3;
  float* d = dist + (size_t)b * Nin;
  int* oi = outIdx + (size_t)b * NPt;
  __shared__ float sv[256];
  __shared__ int   si[256];
  __shared__ float cx, cy, cz;
  int tid = threadIdx.x;
  for (int n = tid; n < Nin; n += 256) d[n] = 1e10f;
  if (tid == 0) oi[0] = 0;
  __syncthreads();
  int cur = 0;
  for (int it = 1; it < NPt; ++it) {
    if (tid == 0) { cx = p[cur * 3]; cy = p[cur * 3 + 1]; cz = p[cur * 3 + 2]; }
    __syncthreads();
    float best = -1.f; int bi = 0x7fffffff;
    for (int n = tid; n < Nin; n += 256) {
      float dx = p[n * 3] - cx, dy = p[n * 3 + 1] - cy, dz = p[n * 3 + 2] - cz;
      float dd = dx * dx + dy * dy + dz * dz;
      float nv = fminf(d[n], dd);
      d[n] = nv;
      if (nv > best || (nv == best && n < bi)) { best = nv; bi = n; }
    }
    sv[tid] = best; si[tid] = bi;
    __syncthreads();
    for (int st = 128; st > 0; st >>= 1) {
      if (tid < st) {
        if (sv[tid + st] > sv[tid] || (sv[tid + st] == sv[tid] && si[tid + st] < si[tid])) {
          sv[tid] = sv[tid + st]; si[tid] = si[tid + st];
        }
      }
      __syncthreads();
    }
    cur = si[0];
    if (tid == 0) oi[it] = cur;
    __syncthreads();
  }
}

__global__ __launch_bounds__(256)
void k_gather_pts(const float* __restrict__ P, const int* __restrict__ fidx,
                  int Nin, int NP, float* __restrict__ C) {
  int i = blockIdx.x * 256 + threadIdx.x;
  if (i >= NB * NP * 3) return;
  int b = i / (NP * 3); int rem = i % (NP * 3); int p = rem / 3; int dd = rem % 3;
  C[i] = P[((size_t)b * Nin + fidx[b * NP + p]) * 3 + dd];
}

// Ball query: first K indices (ascending) with d^2 <= r^2; pad with first.
__global__ __launch_bounds__(256)
void k_ball(const float* __restrict__ P, const float* __restrict__ C,
            int Nin, int NP, int Kk, float r2, int* __restrict__ idx) {
  int t = blockIdx.x * 256 + threadIdx.x;
  if (t >= NB * NP) return;
  int b = t / NP;
  const float* pp = P + (size_t)b * Nin * 3;
  float cx = C[t * 3], cy = C[t * 3 + 1], cz = C[t * 3 + 2];
  int* o = idx + (size_t)t * Kk;
  int cnt = 0, first = 0;
  for (int n = 0; n < Nin && cnt < Kk; ++n) {
    float dx = pp[n * 3] - cx, dy = pp[n * 3 + 1] - cy, dz = pp[n * 3 + 2] - cz;
    float dd = dx * dx + dy * dy + dz * dz;
    if (dd <= r2) { if (cnt == 0) first = n; o[cnt++] = n; }
  }
  for (; cnt < Kk; ++cnt) o[cnt] = first;
}

// Build grouped matrix G[(Cf+3), NB*NP*K]: features gathered, then rel-coords.
__global__ __launch_bounds__(256)
void k_group(const float* __restrict__ F, const float* __restrict__ P,
             const float* __restrict__ C, const int* __restrict__ idx,
             int Nin, int NP, int Kk, int Cf, float* __restrict__ G) {
  long cols = (long)NB * NP * Kk;
  long total = (long)(Cf + 3) * cols;
  long i = (long)blockIdx.x * 256 + threadIdx.x;
  if (i >= total) return;
  int c = (int)(i / cols);
  long col = i % cols;
  int b = (int)(col / ((long)NP * Kk));
  long rem = col % ((long)NP * Kk);
  int p = (int)(rem / Kk);
  int j = idx[col];
  if (c < Cf) {
    G[i] = F[(size_t)c * NB * Nin + (size_t)b * Nin + j];
  } else {
    int dd = c - Cf;
    G[i] = P[((size_t)b * Nin + j) * 3 + dd] - C[((size_t)b * NP + p) * 3 + dd];
  }
}

__global__ __launch_bounds__(256)
void k_maxpool(const float* __restrict__ Y, int Cout, int NPtot, int Kk,
               float* __restrict__ outF, int cOff) {
  int i = blockIdx.x * 256 + threadIdx.x;
  if (i >= Cout * NPtot) return;
  int o = i / NPtot, pc = i % NPtot;
  const float* src = Y + (size_t)o * NPtot * Kk + (size_t)pc * Kk;
  float m = src[0];
  for (int k = 1; k < Kk; ++k) m = fmaxf(m, src[k]);
  outF[(size_t)(cOff + o) * NPtot + pc] = m;
}

// 3-NN with stable (first-index) ties + normalized inverse-distance weights.
__global__ __launch_bounds__(256)
void k_three_nn(const float* __restrict__ P1, const float* __restrict__ P2,
                int n1, int n2, int* __restrict__ nidx, float* __restrict__ w) {
  int t = blockIdx.x * 256 + threadIdx.x;
  if (t >= NB * n1) return;
  int b = t / n1;
  float x = P1[t * 3], y = P1[t * 3 + 1], z = P1[t * 3 + 2];
  const float* q = P2 + (size_t)b * n2 * 3;
  float d0 = 3e38f, d1 = 3e38f, d2 = 3e38f;
  int i0 = 0, i1 = 0, i2 = 0;
  for (int j = 0; j < n2; ++j) {
    float dx = q[j * 3] - x, dy = q[j * 3 + 1] - y, dz = q[j * 3 + 2] - z;
    float dd = dx * dx + dy * dy + dz * dz;
    if (dd < d0)      { d2 = d1; i2 = i1; d1 = d0; i1 = i0; d0 = dd; i0 = j; }
    else if (dd < d1) { d2 = d1; i2 = i1; d1 = dd; i1 = j; }
    else if (dd < d2) { d2 = dd; i2 = j; }
  }
  float w0 = 1.f / (d0 + 1e-8f), w1 = 1.f / (d1 + 1e-8f), w2 = 1.f / (d2 + 1e-8f);
  float s = w0 + w1 + w2;
  nidx[t * 3] = i0; nidx[t * 3 + 1] = i1; nidx[t * 3 + 2] = i2;
  w[t * 3] = w0 / s; w[t * 3 + 1] = w1 / s; w[t * 3 + 2] = w2 / s;
}

// FP input matrix: rows [0,C1) = points1 features, rows [C1,C1+C2) = interpolated.
__global__ __launch_bounds__(256)
void k_fp_build(const float* __restrict__ F1, const float* __restrict__ F2,
                const int* __restrict__ nidx, const float* __restrict__ w,
                int n1, int n2, int C1, int C2, float* __restrict__ Mt) {
  long cols = (long)NB * n1;
  long total = (long)(C1 + C2) * cols;
  long i = (long)blockIdx.x * 256 + threadIdx.x;
  if (i >= total) return;
  int c = (int)(i / cols);
  long col = i % cols;
  if (c < C1) {
    Mt[i] = F1[(size_t)c * cols + col];
  } else {
    int c2 = c - C1;
    int b = (int)(col / n1);
    const float* fr = F2 + (size_t)c2 * NB * n2 + (size_t)b * n2;
    long base = col * 3;
    Mt[i] = w[base] * fr[nidx[base]] + w[base + 1] * fr[nidx[base + 1]] +
            w[base + 2] * fr[nidx[base + 2]];
  }
}

// Head input: 512 feature rows + 3 coordinate rows, 512 columns.
__global__ __launch_bounds__(256)
void k_head_build(const float* __restrict__ F, const float* __restrict__ P,
                  float* __restrict__ Mt) {
  int cols = NB * 256;
  int i = blockIdx.x * 256 + threadIdx.x;
  if (i >= 515 * cols) return;
  int c = i / cols, col = i % cols;
  if (c < 512) Mt[i] = F[(size_t)c * cols + col];
  else         Mt[i] = P[(size_t)col * 3 + (c - 512)];
}

// (C, NB*n) channel-major  ->  (NB, C, n)
__global__ __launch_bounds__(256)
void k_chm_to_bcn(const float* __restrict__ M, int C, int n, float* __restrict__ out) {
  long i = (long)blockIdx.x * 256 + threadIdx.x;
  if (i >= (long)NB * C * n) return;
  int b = (int)(i / ((long)C * n));
  long rem = i % ((long)C * n);
  int c = (int)(rem / n);
  int j = (int)(rem % n);
  out[i] = M[(size_t)c * NB * n + (size_t)b * n + j];
}

__global__ __launch_bounds__(256)
void k_l0xyz_out(const float* __restrict__ xyz, float* __restrict__ out) {
  int i = blockIdx.x * 256 + threadIdx.x;
  if (i >= NB * 3 * N0) return;
  int b = i / (3 * N0); int rem = i % (3 * N0); int c = rem / N0; int n = rem % N0;
  out[i] = xyz[((size_t)b * 6 + c) * N0 + n];
}

// (NB, n, 3) point-major -> (NB, 3, n)
__global__ __launch_bounds__(256)
void k_pts_to_bcn(const float* __restrict__ P, int n, float* __restrict__ out) {
  int i = blockIdx.x * 256 + threadIdx.x;
  if (i >= NB * 3 * n) return;
  int b = i / (3 * n); int rem = i % (3 * n); int c = rem / n; int j = rem % n;
  out[i] = P[((size_t)b * n + j) * 3 + c];
}

// ---------------------------------------------------------------------------
// Host orchestration
// ---------------------------------------------------------------------------
extern "C" void kernel_launch(void* const* d_in, const int* in_sizes, int n_in,
                              void* d_out, int out_size, void* d_ws, size_t ws_size,
                              hipStream_t stream) {
  (void)in_sizes; (void)n_in; (void)out_size; (void)ws_size;
  const float* xyz = (const float*)d_in[0];

  // conv j: W=d_in[1+4j], b=d_in[2+4j], gamma=d_in[3+4j], beta=d_in[4+4j]
  auto Wp = [&](int j) { return (const float*)d_in[1 + 4 * j]; };
  auto bp = [&](int j) { return (const float*)d_in[2 + 4 * j]; };
  auto Gp = [&](int j) { return (const float*)d_in[3 + 4 * j]; };
  auto Bp = [&](int j) { return (const float*)d_in[4 + 4 * j]; };

  // bump allocator over workspace (units: floats, 1KB-aligned chunks)
  float* wsf = (float*)d_ws;
  size_t off = 0;
  auto alloc = [&](size_t nfloats) {
    size_t o = off;
    off += (nfloats + 255) & ~(size_t)255;
    return wsf + o;
  };

  float* l0P   = alloc((size_t)NB * N0 * 3);
  float* l1P   = alloc((size_t)NB * 1024 * 3);
  float* l2P   = alloc((size_t)NB * 512 * 3);
  float* l3P   = alloc((size_t)NB * 256 * 3);
  float* l0F   = alloc((size_t)6 * NB * N0);
  float* l1F   = alloc((size_t)64 * NB * 1024);
  float* l2F   = alloc((size_t)256 * NB * 512);
  float* l3F   = alloc((size_t)512 * NB * 256);
  float* l2F2  = alloc((size_t)256 * NB * 512);
  float* l1F2  = alloc((size_t)128 * NB * 1024);
  float* l0out = alloc((size_t)32 * NB * N0);
  float* meanB = alloc(512);
  float* rstdB = alloc(512);
  float* fpsD  = alloc((size_t)NB * N0);
  int*   fpsI  = (int*)alloc((size_t)NB * 1024);
  int*   ballI = (int*)alloc((size_t)NB * 1024 * 64);
  int*   nnI   = (int*)alloc((size_t)NB * N0 * 3);
  float* nnW   = alloc((size_t)NB * N0 * 3);
  float* headO = alloc((size_t)3 * NB * 256);
  float* headD = alloc((size_t)1 * NB * 256);
  float* Gbuf  = alloc((size_t)262 * NB * 16384);   // max (259+3) x 32768
  float* Y0b   = alloc((size_t)196 * NB * 16384);   // max 196 x 32768
  float* Y1b   = alloc((size_t)256 * NB * 16384);   // max 256 x 32768 (== 128 x 65536)

  auto gemm = [&](const float* A, const float* X, float* Y, int M, int K, int N,
                  const float* bias) {
    dim3 g((unsigned)((N + 15) / 16), (unsigned)((M + 127) / 128));
    k_gemm<<<g, 256, 0, stream>>>(A, X, Y, M, K, N, bias);
  };
  auto bn = [&](float* Y, int M, int N, const float* gamma, const float* beta) {
    k_rowstats<<<M, 256, 0, stream>>>(Y, N, meanB, rstdB);
    long total = (long)M * N;
    k_bn_relu<<<(unsigned)((total + 255) / 256), 256, 0, stream>>>(Y, total, N, meanB,
                                                                   rstdB, gamma, beta);
  };

  auto run_sa = [&](const float* P, const float* F, int Cf, int Nin, int NP,
                    float* newP, int convBase, const int* Ks, const float* R2,
                    const int* C0, const int* C1, float* outF) {
    k_fps<<<NB, 256, 0, stream>>>(P, Nin, NP, fpsD, fpsI);
    k_gather_pts<<<(NB * NP * 3 + 255) / 256, 256, 0, stream>>>(P, fpsI, Nin, NP, newP);
    int cOff = 0;
    for (int br = 0; br < 2; ++br) {
      int Kk = Ks[br];
      long cols = (long)NB * NP * Kk;
      int Cin = Cf + 3;
      k_ball<<<(NB * NP + 255) / 256, 256, 0, stream>>>(P, newP, Nin, NP, Kk, R2[br], ballI);
      long gt = (long)Cin * cols;
      k_group<<<(unsigned)((gt + 255) / 256), 256, 0, stream>>>(F, P, newP, ballI,
                                                                Nin, NP, Kk, Cf, Gbuf);
      int j0 = convBase + 2 * br, j1 = j0 + 1;
      gemm(Wp(j0), Gbuf, Y0b, C0[br], Cin, (int)cols, nullptr);
      bn(Y0b, C0[br], (int)cols, Gp(j0), Bp(j0));
      gemm(Wp(j1), Y0b, Y1b, C1[br], C0[br], (int)cols, nullptr);
      bn(Y1b, C1[br], (int)cols, Gp(j1), Bp(j1));
      int NPtot = NB * NP;
      k_maxpool<<<(C1[br] * NPtot + 255) / 256, 256, 0, stream>>>(Y1b, C1[br], NPtot,
                                                                  Kk, outF, cOff);
      cOff += C1[br];
    }
  };

  auto run_fp = [&](const float* P1, const float* P2, int n1, int n2,
                    const float* F1, int C1, const float* F2, int C2,
                    int convBase, int Cm0, int Cm1, float* outF) {
    k_three_nn<<<(NB * n1 + 255) / 256, 256, 0, stream>>>(P1, P2, n1, n2, nnI, nnW);
    int cols = NB * n1, Cin = C1 + C2;
    long bt = (long)Cin * cols;
    k_fp_build<<<(unsigned)((bt + 255) / 256), 256, 0, stream>>>(F1, F2, nnI, nnW,
                                                                 n1, n2, C1, C2, Gbuf);
    int j0 = convBase, j1 = convBase + 1;
    gemm(Wp(j0), Gbuf, Y0b, Cm0, Cin, cols, nullptr);
    bn(Y0b, Cm0, cols, Gp(j0), Bp(j0));
    gemm(Wp(j1), Y0b, outF, Cm1, Cm0, cols, nullptr);
    bn(outF, Cm1, cols, Gp(j1), Bp(j1));
  };

  // ------------------- pipeline -------------------
  k_init_l0<<<(NB * 6 * N0 + 255) / 256, 256, 0, stream>>>(xyz, l0P, l0F);

  const int   Ks[2] = {32, 64};
  const int   C32[2] = {32, 32},  C64[2] = {64, 64};
  const int   C128[2] = {128, 128}, C196[2] = {196, 196}, C256[2] = {256, 256};
  const float R1[2] = {0.025f * 0.025f, 0.05f * 0.05f};
  const float R2s[2] = {0.05f * 0.05f, 0.1f * 0.1f};
  const float R3[2] = {0.1f * 0.1f, 0.2f * 0.2f};

  run_sa(l0P, l0F,   6, N0,   1024, l1P, 0, Ks, R1,  C32,  C32,  l1F);
  run_sa(l1P, l1F,  64, 1024,  512, l2P, 4, Ks, R2s, C64,  C128, l2F);
  run_sa(l2P, l2F, 256, 512,   256, l3P, 8, Ks, R3,  C196, C256, l3F);

  run_fp(l2P, l3P, 512,  256, l2F,  256, l3F,  512, 12, 256, 256, l2F2);
  run_fp(l1P, l2P, 1024, 512, l1F,   64, l2F2, 256, 14, 128, 128, l1F2);
  run_fp(l0P, l1P, N0,  1024, l0F,    6, l1F2, 128, 16,  64,  32, l0out);

  // heads on l3 (515 channels x 512 cols)
  k_head_build<<<(515 * NB * 256 + 255) / 256, 256, 0, stream>>>(l3F, l3P, Gbuf);
  gemm(Wp(18), Gbuf, Y0b, 256, 515, NB * 256, nullptr);
  bn(Y0b, 256, NB * 256, Gp(18), Bp(18));
  gemm(Wp(19), Y0b, headO, 3, 256, NB * 256, bp(19));
  gemm(Wp(20), Gbuf, Y1b, 256, 515, NB * 256, nullptr);
  bn(Y1b, 256, NB * 256, Gp(20), Bp(20));
  gemm(Wp(21), Y1b, headD, 1, 256, NB * 256, bp(21));

  // ------------------- outputs -------------------
  float* out = (float*)d_out;
  size_t o = 0;
  k_chm_to_bcn<<<(NB * 32 * N0 + 255) / 256, 256, 0, stream>>>(l0out, 32, N0, out + o);
  o += (size_t)NB * 32 * N0;
  k_chm_to_bcn<<<(NB * 512 * 256 + 255) / 256, 256, 0, stream>>>(l3F, 512, 256, out + o);
  o += (size_t)NB * 512 * 256;
  k_l0xyz_out<<<(NB * 3 * N0 + 255) / 256, 256, 0, stream>>>(xyz, out + o);
  o += (size_t)NB * 3 * N0;
  k_pts_to_bcn<<<(NB * 3 * 256 + 255) / 256, 256, 0, stream>>>(l3P, 256, out + o);
  o += (size_t)NB * 3 * 256;
  k_chm_to_bcn<<<(NB * 3 * 256 + 255) / 256, 256, 0, stream>>>(headO, 3, 256, out + o);
  o += (size_t)NB * 3 * 256;
  k_chm_to_bcn<<<(NB * 1 * 256 + 255) / 256, 256, 0, stream>>>(headD, 1, 256, out + o);
}
